// MyAttention_87299505258745
// MI455X (gfx1250) — matrix operand order repaired and checked by reading, not compiled, not run on hardware
//
#include <hip/hip_runtime.h>

#define DIM      768
#define HEADS    12
#define DH       64
#define NTOK     4096
#define BM       64
#define BK       64
#define LDA      72      // bf16 LDS row stride (pad, keeps 16B alignment)
#define LDC      68      // fp32 LDS row stride
#define NSPLIT   8
#define LN_EPS   1e-5f
#define INVN     (1.0f / 4096.0f)

typedef __attribute__((ext_vector_type(16))) __bf16 bf16x16;
typedef __attribute__((ext_vector_type(8)))  __bf16 bf16x8;
typedef __attribute__((ext_vector_type(8)))  float  f32x8;
typedef __attribute__((ext_vector_type(8)))  unsigned short us8;

static __device__ __forceinline__ unsigned short f2bf(float f) {
  unsigned int u = __float_as_uint(f);
  u += 0x7FFFu + ((u >> 16) & 1u);          // round-to-nearest-even
  return (unsigned short)(u >> 16);
}
static __device__ __forceinline__ __bf16 f2bfh(float f) {
  unsigned short s = f2bf(f);
  return __builtin_bit_cast(__bf16, s);
}
static __device__ __forceinline__ void cvt_store4(unsigned short* dst, float4 v) {
  ushort4 o;
  o.x = f2bf(v.x); o.y = f2bf(v.y); o.z = f2bf(v.z); o.w = f2bf(v.w);
  *(ushort4*)dst = o;
}

// A fragment 16x32 bf16 from row-major [row][ld]; contiguous k order used here (same as B):
// lanes 0-15 hold K 0..15, lanes 16-31 hold K 16..31.
static __device__ __forceinline__ bf16x16 load_frag_a(const __bf16* p, int ld,
                                                      int row0, int k0, int lane) {
  const int hf = lane >> 4, m = lane & 15;
  const __bf16* q = p + (row0 + m) * ld + k0 + hf * 16;
  bf16x8 lo = *(const bf16x8*)(q);
  bf16x8 hi = *(const bf16x8*)(q + 8);
  bf16x16 r;
#pragma unroll
  for (int i = 0; i < 8; ++i) { r[i] = lo[i]; r[i + 8] = hi[i]; }
  return r;
}
// B fragment 32x16 bf16 from "BT" storage [n][ld] (element (k,n) at p[n*ld+k]);
// lanes 0-15 hold K 0..15, lanes 16-31 hold K 16..31.
static __device__ __forceinline__ bf16x16 load_frag_b(const __bf16* p, int ld,
                                                      int n0, int k0, int lane) {
  const int hf = lane >> 4, n = lane & 15;
  const __bf16* q = p + (n0 + n) * ld + k0 + hf * 16;
  bf16x8 lo = *(const bf16x8*)(q);
  bf16x8 hi = *(const bf16x8*)(q + 8);
  bf16x16 r;
#pragma unroll
  for (int i = 0; i < 8; ++i) { r[i] = lo[i]; r[i + 8] = hi[i]; }
  return r;
}
// Same B fragment, but source is fp32 (converted on the fly).
static __device__ __forceinline__ bf16x16 load_frag_b_f32(const float* p, int ld,
                                                          int n0, int k0, int lane) {
  const int hf = lane >> 4, n = lane & 15;
  const float* q = p + (n0 + n) * ld + k0 + hf * 16;
  bf16x16 r;
#pragma unroll
  for (int i = 0; i < 16; ++i) r[i] = f2bfh(q[i]);
  return r;
}
// C layout: lane<16 -> col=lane, rows r; lane>=16 -> col=lane-16, rows r+8.
static __device__ __forceinline__ void store_c_f32(float* cs, int ld, int row0,
                                                   int col0, int lane, f32x8 a) {
  const int hf = lane >> 4, n = lane & 15;
  const int r0 = row0 + hf * 8;
#pragma unroll
  for (int r = 0; r < 8; ++r) cs[(r0 + r) * ld + col0 + n] = a[r];
}
static __device__ __forceinline__ void store_c_bf16(unsigned short* cs, int ld, int row0,
                                                    int col0, int lane, f32x8 a) {
  const int hf = lane >> 4, n = lane & 15;
  const int r0 = row0 + hf * 8;
#pragma unroll
  for (int r = 0; r < 8; ++r) cs[(r0 + r) * ld + col0 + n] = f2bf(a[r]);
}

#define WMMA_BF16(A, B, C) \
  __builtin_amdgcn_wmma_f32_16x16x32_bf16(false, (A), false, (B), (short)0, (C), false, false)

// Async global->LDS 16B copy (CDNA5 GLOBAL_LOAD_ASYNC_TO_LDS_B128, ASYNCcnt).
// Only valid when src/dst are 16B aligned. Fallback: VGPR round-trip copy.
#if defined(__gfx1250__)
#define ASYNC_CP16(lds_dst, g_src)                                              \
  do {                                                                          \
    unsigned _lo = (unsigned)(size_t)(const void*)(lds_dst);                    \
    unsigned long long _ga = (unsigned long long)(size_t)(const void*)(g_src);  \
    asm volatile("global_load_async_to_lds_b128 %0, %1, off"                    \
                 :: "v"(_lo), "v"(_ga) : "memory");                             \
  } while (0)
#define ASYNC_WAIT() asm volatile("s_wait_asynccnt 0x0" ::: "memory")
#else
#define ASYNC_CP16(lds_dst, g_src) (*(us8*)(lds_dst) = *(const us8*)(g_src))
#define ASYNC_WAIT()
#endif

// ---------------------------------------------------------------------------
// Kernel 1: qkv = x @ w_map^T (bf16 WMMA, fp32 acc). q -> bf16 scratch.
// k,v -> fp32 LayerNorm over dim_head (k scaled by 1/n), stored TRANSPOSED
// bf16 as [bh][d][n] so the context GEMM gets K-contiguous fragments.
// grid (512 token blocks, 36 feature blocks), 256 threads (8 waves).
// ---------------------------------------------------------------------------
__global__ __launch_bounds__(256) void k_qkv(
    const float* __restrict__ x, const float* __restrict__ w_map,
    const float* __restrict__ g_k, const float* __restrict__ b_k,
    const float* __restrict__ g_v, const float* __restrict__ b_v,
    unsigned short* __restrict__ qbuf, unsigned short* __restrict__ kT,
    unsigned short* __restrict__ vT) {
  __shared__ unsigned short xs[BM * LDA];
  __shared__ unsigned short wt[BM * LDA];
  __shared__ float          cs[BM * LDC];
  __shared__ unsigned short ns[BM * LDA];

  const int tb = blockIdx.x, nb = blockIdx.y;
  const int tid = threadIdx.x, lane = tid & 31, wv = tid >> 5;
  const int mi = wv & 3, ni0 = (wv >> 2) * 2, ni1 = ni0 + 1;
  const long tok0 = (long)tb * BM;

  f32x8 acc0 = {}; f32x8 acc1 = {};
  const __bf16* xsb = (const __bf16*)xs;
  const __bf16* wtb = (const __bf16*)wt;

  for (int ks = 0; ks < DIM; ks += BK) {
    __syncthreads();
#pragma unroll
    for (int i = 0; i < 4; ++i) {           // stage x[64][64] and w[64][64] as bf16
      int lin = tid + i * 256;
      int r = lin >> 4, c = (lin & 15) * 4;
      float4 xv = *(const float4*)(x + (tok0 + r) * DIM + ks + c);
      cvt_store4(xs + r * LDA + c, xv);
      float4 wv4 = *(const float4*)(w_map + ((long)nb * BM + r) * DIM + ks + c);
      cvt_store4(wt + r * LDA + c, wv4);
    }
    __syncthreads();
#pragma unroll
    for (int kc = 0; kc < BK; kc += 32) {
      bf16x16 a  = load_frag_a(xsb, LDA, mi * 16, kc, lane);
      bf16x16 b0 = load_frag_b(wtb, LDA, ni0 * 16, kc, lane);
      bf16x16 b1 = load_frag_b(wtb, LDA, ni1 * 16, kc, lane);
      acc0 = WMMA_BF16(a, b0, acc0);
      acc1 = WMMA_BF16(a, b1, acc1);
    }
  }
  __syncthreads();

  if (nb < HEADS) {                          // ---- q path ----
    store_c_bf16(ns, LDA, mi * 16, ni0 * 16, lane, acc0);
    store_c_bf16(ns, LDA, mi * 16, ni1 * 16, lane, acc1);
    __syncthreads();
    int t = tid & 63, c = (tid >> 6) * 16;
    unsigned short* dst = qbuf + (tok0 + t) * DIM + nb * DH + c;
    *(us8*)(dst)     = *(const us8*)(ns + t * LDA + c);
    *(us8*)(dst + 8) = *(const us8*)(ns + t * LDA + c + 8);
  } else {                                   // ---- k / v path ----
    const bool isK = nb < 2 * HEADS;
    const int h = nb - (isK ? HEADS : 2 * HEADS);
    store_c_f32(cs, LDC, mi * 16, ni0 * 16, lane, acc0);
    store_c_f32(cs, LDC, mi * 16, ni1 * 16, lane, acc1);
    __syncthreads();
    if (tid < 64) {                          // per-token LayerNorm over 64 features
      const float* g  = isK ? g_k : g_v;
      const float* be = isK ? b_k : b_v;
      const float* row = cs + tid * LDC;
      float s1 = 0.f, s2 = 0.f;
      for (int j = 0; j < DH; ++j) { float v = row[j]; s1 += v; s2 += v * v; }
      float mu  = s1 * (1.0f / DH);
      float var = s2 * (1.0f / DH) - mu * mu;
      float rr  = rsqrtf(var + LN_EPS);
      float sc  = isK ? INVN : 1.0f;         // fold 1/n into k
      for (int j = 0; j < DH; ++j) {
        float v = ((row[j] - mu) * rr * g[j] + be[j]) * sc;
        ns[tid * LDA + j] = f2bf(v);
      }
    }
    __syncthreads();
    if (tid < 64) {                          // transposed store: [bh][d][n], 128B rows
      int d = tid;
      long bh = (long)(tb >> 6) * HEADS + h;
      unsigned short* dst = (isK ? kT : vT) + (bh * DH + d) * (long)NTOK + (long)(tb & 63) * 64;
#pragma unroll
      for (int c = 0; c < 64; c += 8) {
        us8 v;
#pragma unroll
        for (int j = 0; j < 8; ++j) v[j] = ns[(c + j) * LDA + d];
        *(us8*)(dst + c) = v;
      }
    }
  }
}

// ---------------------------------------------------------------------------
// Kernel 2: per-(b,h) context partials. Computes C'[e][d] = sum_t v[t,e]k[t,d]
// (= ctx^T) via WMMA with direct global fragment loads (transposed layout is
// K-contiguous). Token range split 8 ways -> deterministic tree reduction.
// grid (96, 8), 256 threads.
// ---------------------------------------------------------------------------
__global__ __launch_bounds__(256) void k_ctx(
    const unsigned short* __restrict__ kT, const unsigned short* __restrict__ vT,
    float* __restrict__ part) {
  const int bh = blockIdx.x, sp = blockIdx.y;
  const int tid = threadIdx.x, lane = tid & 31, wv = tid >> 5;
  const int mi = wv & 3, ni0 = (wv >> 2) * 2, ni1 = ni0 + 1;
  const __bf16* kb = (const __bf16*)kT + (long)bh * DH * NTOK;
  const __bf16* vb = (const __bf16*)vT + (long)bh * DH * NTOK;
  f32x8 acc0 = {}; f32x8 acc1 = {};
  const int t0 = sp * (NTOK / NSPLIT);
  for (int t = t0; t < t0 + NTOK / NSPLIT; t += 32) {
    bf16x16 a  = load_frag_a(vb, NTOK, mi * 16, t, lane);   // A[e][t] = v^T
    bf16x16 b0 = load_frag_b(kb, NTOK, ni0 * 16, t, lane);  // B[t][d] = k
    bf16x16 b1 = load_frag_b(kb, NTOK, ni1 * 16, t, lane);
    acc0 = WMMA_BF16(a, b0, acc0);
    acc1 = WMMA_BF16(a, b1, acc1);
  }
  float* dst = part + ((long)sp * (HEADS * 8) + bh) * (DH * DH);
  const int hf = lane >> 4, n = lane & 15;
  const int r0 = mi * 16 + hf * 8;
#pragma unroll
  for (int r = 0; r < 8; ++r) {
    dst[(r0 + r) * DH + ni0 * 16 + n] = acc0[r];
    dst[(r0 + r) * DH + ni1 * 16 + n] = acc1[r];
  }
}

__global__ void k_ctx_reduce(const float* __restrict__ part, float* __restrict__ ctxT) {
  const int n = 96 * DH * DH;
  int i = blockIdx.x * 256 + threadIdx.x;
  if (i < n) {
    float s = 0.f;
#pragma unroll
    for (int sp = 0; sp < NSPLIT; ++sp) s += part[(long)sp * n + i];
    ctxT[i] = s;
  }
}

// ---------------------------------------------------------------------------
// Kernel 3: y[:, h*64+e] = q_head @ ctx. A frags direct from bf16 q (K-contig),
// B frags from fp32 ctx^T (K-contig), converted to bf16 at load. y -> bf16.
// grid (512, 12), 256 threads.
// ---------------------------------------------------------------------------
__global__ __launch_bounds__(256) void k_qctx(
    const unsigned short* __restrict__ qbuf, const float* __restrict__ ctxT,
    unsigned short* __restrict__ ybuf) {
  __shared__ unsigned short ys[BM * LDA];
  const int tb = blockIdx.x, h = blockIdx.y;
  const int tid = threadIdx.x, lane = tid & 31, wv = tid >> 5;
  const int mi = wv & 3, ni0 = (wv >> 2) * 2, ni1 = ni0 + 1;
  const long tok0 = (long)tb * BM;
  const int bh = (tb >> 6) * HEADS + h;
  const __bf16* qb = (const __bf16*)qbuf + tok0 * DIM + h * DH;
  const float* cb = ctxT + (long)bh * DH * DH;     // ctxT[e][d]; B[k=d][n=e]
  f32x8 acc0 = {}; f32x8 acc1 = {};
#pragma unroll
  for (int kc = 0; kc < DH; kc += 32) {
    bf16x16 a  = load_frag_a(qb, DIM, mi * 16, kc, lane);
    bf16x16 b0 = load_frag_b_f32(cb, DH, ni0 * 16, kc, lane);
    bf16x16 b1 = load_frag_b_f32(cb, DH, ni1 * 16, kc, lane);
    acc0 = WMMA_BF16(a, b0, acc0);
    acc1 = WMMA_BF16(a, b1, acc1);
  }
  store_c_bf16(ys, LDA, mi * 16, ni0 * 16, lane, acc0);
  store_c_bf16(ys, LDA, mi * 16, ni1 * 16, lane, acc1);
  __syncthreads();
  int t = tid & 63, c = (tid >> 6) * 16;
  unsigned short* dst = ybuf + (tok0 + t) * DIM + h * DH + c;
  *(us8*)(dst)     = *(const us8*)(ys + t * LDA + c);
  *(us8*)(dst + 8) = *(const us8*)(ys + t * LDA + c + 8);
}

// ---------------------------------------------------------------------------
// Kernel 4: out = y @ w_out^T + b_out (fp32 out). y tile staged with async
// global->LDS B128 (no conversion needed), w tile via VGPR cvt path.
// grid (512, 12), 256 threads.
// ---------------------------------------------------------------------------
__global__ __launch_bounds__(256) void k_out(
    const unsigned short* __restrict__ ybuf, const float* __restrict__ w_out,
    const float* __restrict__ b_out, float* __restrict__ out) {
  __shared__ unsigned short ysd[BM * LDA];
  __shared__ unsigned short wts[BM * LDA];
  const int tb = blockIdx.x, nb = blockIdx.y;
  const int tid = threadIdx.x, lane = tid & 31, wv = tid >> 5;
  const int mi = wv & 3, ni0 = (wv >> 2) * 2, ni1 = ni0 + 1;
  const long tok0 = (long)tb * BM;
  f32x8 acc0 = {}; f32x8 acc1 = {};

  for (int ks = 0; ks < DIM; ks += BK) {
    __syncthreads();
#pragma unroll
    for (int i = 0; i < 2; ++i) {            // y tile already bf16: async 16B copies
      int lin = tid + i * 256;
      int r = lin >> 3, c = (lin & 7) * 8;
      ASYNC_CP16(ysd + r * LDA + c, ybuf + (tok0 + r) * DIM + ks + c);
    }
#pragma unroll
    for (int i = 0; i < 4; ++i) {            // w_out tile fp32 -> bf16
      int lin = tid + i * 256;
      int r = lin >> 4, c = (lin & 15) * 4;
      float4 wv4 = *(const float4*)(w_out + ((long)nb * BM + r) * DIM + ks + c);
      cvt_store4(wts + r * LDA + c, wv4);
    }
    ASYNC_WAIT();
    __syncthreads();
#pragma unroll
    for (int kc = 0; kc < BK; kc += 32) {
      bf16x16 a  = load_frag_a((const __bf16*)ysd, LDA, mi * 16, kc, lane);
      bf16x16 b0 = load_frag_b((const __bf16*)wts, LDA, ni0 * 16, kc, lane);
      bf16x16 b1 = load_frag_b((const __bf16*)wts, LDA, ni1 * 16, kc, lane);
      acc0 = WMMA_BF16(a, b0, acc0);
      acc1 = WMMA_BF16(a, b1, acc1);
    }
  }
  const int hf = lane >> 4, n = lane & 15;
  const int col0 = nb * BM + ni0 * 16 + n;
  const int col1 = nb * BM + ni1 * 16 + n;
  const float bb0 = b_out[col0], bb1 = b_out[col1];
  const int r0 = mi * 16 + hf * 8;
#pragma unroll
  for (int r = 0; r < 8; ++r) {
    out[(tok0 + r0 + r) * DIM + col0] = acc0[r] + bb0;
    out[(tok0 + r0 + r) * DIM + col1] = acc1[r] + bb1;
  }
}

// ---------------------------------------------------------------------------
extern "C" void kernel_launch(void* const* d_in, const int* in_sizes, int n_in,
                              void* d_out, int out_size, void* d_ws, size_t ws_size,
                              hipStream_t stream) {
  (void)in_sizes; (void)n_in; (void)out_size; (void)ws_size;
  const float* x     = (const float*)d_in[0];
  const float* w_map = (const float*)d_in[1];
  const float* g_k   = (const float*)d_in[2];
  const float* b_k   = (const float*)d_in[3];
  const float* g_v   = (const float*)d_in[4];
  const float* b_v   = (const float*)d_in[5];
  const float* w_out = (const float*)d_in[6];
  const float* b_out = (const float*)d_in[7];
  float* out = (float*)d_out;

  char* ws = (char*)d_ws;
  const size_t SZ_BF = (size_t)8 * NTOK * DIM * 2;        // 50.3 MB per bf16 tensor
  unsigned short* qb = (unsigned short*)(ws);
  unsigned short* kT = (unsigned short*)(ws + SZ_BF);
  unsigned short* vT = (unsigned short*)(ws + 2 * SZ_BF);
  float* ctx_part    = (float*)(ws + 3 * SZ_BF);          // [8][96][64][64]
  float* ctxT        = (float*)(ws + 3 * SZ_BF + (size_t)NSPLIT * 96 * DH * DH * 4);
  unsigned short* yb = kT;   // kT is dead after k_ctx; reuse (stream-ordered)

  k_qkv<<<dim3(512, 36), dim3(256), 0, stream>>>(x, w_map, g_k, b_k, g_v, b_v, qb, kT, vT);
  k_ctx<<<dim3(96, NSPLIT), dim3(256), 0, stream>>>(kT, vT, ctx_part);
  k_ctx_reduce<<<dim3((96 * DH * DH + 255) / 256), dim3(256), 0, stream>>>(ctx_part, ctxT);
  k_qctx<<<dim3(512, HEADS), dim3(256), 0, stream>>>(qb, ctxT, yb);
  k_out<<<dim3(512, HEADS), dim3(256), 0, stream>>>(yb, w_out, b_out, out);
}
